// Generator_70944269796226
// MI455X (gfx1250) — compile-verified, matrix operand order checked by reading, not run on hardware
//
#include <hip/hip_runtime.h>
#include <hip/hip_bf16.h>
#include <math.h>

typedef _Float16 v8h  __attribute__((ext_vector_type(8)));
typedef _Float16 v16h __attribute__((ext_vector_type(16)));
typedef float    v8f  __attribute__((ext_vector_type(8)));
typedef int      v4i_vs __attribute__((vector_size(16)));   // matches builtin param type

#define LDSTRIDE 72   // 64 halves payload + 8 pad; all fragment offsets stay 16B-aligned

#if defined(__has_builtin)
#if __has_builtin(__builtin_amdgcn_global_load_async_to_lds_b128)
#define USE_ASYNC_LDS 1
#endif
#endif
#ifndef USE_ASYNC_LDS
#define USE_ASYNC_LDS 0
#endif

#if USE_ASYNC_LDS
#define ASYNC_GPTR(p) ((__attribute__((address_space(1))) v4i_vs*)(p))
#define ASYNC_LPTR(p) ((__attribute__((address_space(3))) v4i_vs*)(p))
#endif

__device__ __forceinline__ void wait_asynccnt0() {
#if USE_ASYNC_LDS
#if __has_builtin(__builtin_amdgcn_s_wait_asynccnt)
  __builtin_amdgcn_s_wait_asynccnt(0);
#else
  asm volatile("s_wait_asynccnt 0" ::: "memory");
#endif
#endif
}

// ---------------------------------------------------------------------------
// Implicit-GEMM convolution, f16 operands / f32 accum.
//   M = OC (64/block), N = OH*OW (64/block), K = IC*KS*KS in steps of 64.
//   Weights pre-converted to f16 with rows padded to KtotP (mult of 64) and
//   OC padded to mult of 64 rows (zero fill) -> A staging is an unconditional
//   aligned 16B/lane copy (async-to-LDS when available).
//   8 waves: 4 M-tiles x 2 N-tile pairs; 4x v_wmma_f32_16x16x32_f16 per step.
//   act: 0=none, 1=sigmoid, 2=tanh.  'pad' = virtual zero-padding of input.
// ---------------------------------------------------------------------------
template <int KS>
__global__ __launch_bounds__(256) void conv_wmma_t(
    const float* __restrict__ in, const _Float16* __restrict__ w,
    const float* __restrict__ bias, float* __restrict__ out,
    int IC, int IH, int IW, int OC, int OH, int OW,
    int KtotP, int stride, int pad, int dil, int act)
{
  __shared__ _Float16 ldsA[64 * LDSTRIDE];
  __shared__ _Float16 ldsB[64 * LDSTRIDE];

  constexpr int KK = KS * KS;
  const int Ktot = IC * KK;
  const int OHW  = OH * OW;
  const int tid  = threadIdx.x;
  const int lane = tid & 31;
  const int wv   = tid >> 5;
  const int wm   = wv >> 1;     // 0..3 : M sub-tile
  const int wn   = wv & 1;      // 0..1 : N sub-tile pair
  const int ocBase = blockIdx.x * 64;
  const int nBase  = blockIdx.y * 64;

  v8f acc0, acc1;
  #pragma unroll
  for (int j = 0; j < 8; ++j) { acc0[j] = 0.f; acc1[j] = 0.f; }

  // staging assignment: each thread owns 16 halves: row 0..63, colBase {0,16,32,48}
  const int eRow  = tid >> 2;
  const int eColB = (tid & 3) * 16;

  // per-thread B gather geometry (fixed across K loop)
  const int nIdx = nBase + eRow;
  int oh = 0, ow = 0, ihb = 0, iwb = 0;
  const bool nOK = nIdx < OHW;
  if (nOK) {
    oh = nIdx / OW; ow = nIdx - oh * OW;
    ihb = oh * stride - pad;
    iwb = ow * stride - pad;
  }
  const _Float16* wrow = w + (size_t)(ocBase + eRow) * KtotP + eColB;

  for (int kk0 = 0; kk0 < Ktot; kk0 += 64) {
    // ---- gather B (im2col patches, virtual zero pad) into registers ----
    v16h bvv;
    #pragma unroll
    for (int j = 0; j < 16; ++j) bvv[j] = (_Float16)0.f;
    if (nOK) {
      #pragma unroll
      for (int j = 0; j < 16; ++j) {
        int k = kk0 + eColB + j;
        if (k < Ktot) {
          int ic = k / KK;                 // KK is constexpr -> mul/shift
          int r  = k - ic * KK;
          int ky = r / KS;
          int kx = r - ky * KS;
          int ih = ihb + ky * dil;
          int iw = iwb + kx * dil;
          if ((unsigned)ih < (unsigned)IH && (unsigned)iw < (unsigned)IW)
            bvv[j] = (_Float16)in[((size_t)ic * IH + ih) * IW + iw];
        }
      }
    }
    __syncthreads();                       // prior-iteration LDS reads done
    // ---- stage A (f16 weights, aligned, zero-padded -> unconditional) ----
#if USE_ASYNC_LDS
    __builtin_amdgcn_global_load_async_to_lds_b128(
        ASYNC_GPTR(wrow + kk0),
        ASYNC_LPTR(&ldsA[eRow * LDSTRIDE + eColB]), 0, 0);
    __builtin_amdgcn_global_load_async_to_lds_b128(
        ASYNC_GPTR(wrow + kk0 + 8),
        ASYNC_LPTR(&ldsA[eRow * LDSTRIDE + eColB + 8]), 0, 0);
#else
    {
      v16h av = *(const v16h*)(wrow + kk0);
      *(v16h*)(&ldsA[eRow * LDSTRIDE + eColB]) = av;
    }
#endif
    *(v16h*)(&ldsB[eRow * LDSTRIDE + eColB]) = bvv;
    wait_asynccnt0();
    __syncthreads();

    // ---- two 32-wide K slabs -> 4 WMMA ----
    const int K0    = (lane >= 16) ? 8 : 0;
    const int mRow  = wm * 16 + (lane & 15);
    const int nRow0 = wn * 32 + (lane & 15);
    #pragma unroll
    for (int c = 0; c < 2; ++c) {
      const int co = c * 32 + K0;
      v8h alo  = *(const v8h*)(&ldsA[mRow * LDSTRIDE + co]);
      v8h ahi  = *(const v8h*)(&ldsA[mRow * LDSTRIDE + co + 16]);
      v8h b0lo = *(const v8h*)(&ldsB[nRow0 * LDSTRIDE + co]);
      v8h b0hi = *(const v8h*)(&ldsB[nRow0 * LDSTRIDE + co + 16]);
      v8h b1lo = *(const v8h*)(&ldsB[(nRow0 + 16) * LDSTRIDE + co]);
      v8h b1hi = *(const v8h*)(&ldsB[(nRow0 + 16) * LDSTRIDE + co + 16]);
      v16h a  = __builtin_shufflevector(alo,  ahi,  0,1,2,3,4,5,6,7,8,9,10,11,12,13,14,15);
      v16h b0 = __builtin_shufflevector(b0lo, b0hi, 0,1,2,3,4,5,6,7,8,9,10,11,12,13,14,15);
      v16h b1 = __builtin_shufflevector(b1lo, b1hi, 0,1,2,3,4,5,6,7,8,9,10,11,12,13,14,15);
      acc0 = __builtin_amdgcn_wmma_f32_16x16x32_f16(false, a, false, b0, (short)0, acc0, false, false);
      acc1 = __builtin_amdgcn_wmma_f32_16x16x32_f16(false, a, false, b1, (short)0, acc1, false, false);
    }
  }

  // ---- epilogue: C/D layout -> global, fused bias + activation ----
  {
    int rowAdd = (lane >= 16) ? 8 : 0;
    int n0 = nBase + wn * 32 + (lane & 15);
    int n1 = n0 + 16;
    #pragma unroll
    for (int r = 0; r < 8; ++r) {
      int oc = ocBase + wm * 16 + rowAdd + r;
      if (oc < OC) {
        float bv = bias[oc];
        float o0 = acc0[r] + bv;
        float o1 = acc1[r] + bv;
        if (act == 1)      { o0 = 1.f/(1.f+expf(-o0)); o1 = 1.f/(1.f+expf(-o1)); }
        else if (act == 2) { o0 = tanhf(o0);           o1 = tanhf(o1); }
        if (n0 < OHW) out[(size_t)oc * OHW + n0] = o0;
        if (n1 < OHW) out[(size_t)oc * OHW + n1] = o1;
      }
    }
  }
}

// ---------------------------------------------------------------------------
// fp32 -> f16 weight repack with zero padding: rows OC->OCP, cols Ktot->KtotP
// ---------------------------------------------------------------------------
__global__ __launch_bounds__(256) void cvt_w(
    const float* __restrict__ src, _Float16* __restrict__ dst,
    int OC, int Ktot, int KtotP, long totalP)
{
  for (long i = blockIdx.x * 256L + threadIdx.x; i < totalP; i += (long)gridDim.x * 256L) {
    long row = i / KtotP;
    int  col = (int)(i - row * KtotP);
    float v = (row < OC && col < Ktot) ? src[row * (long)Ktot + col] : 0.f;
    dst[i] = (_Float16)v;
  }
}

// ---------------------------------------------------------------------------
// InstanceNorm (biased var) + LeakyReLU(0.2); one block per channel; in-place ok
// ---------------------------------------------------------------------------
__global__ __launch_bounds__(256) void inorm_lrelu(
    const float* __restrict__ in, float* __restrict__ out, int HW)
{
  const int c = blockIdx.x;
  const float* p = in  + (size_t)c * HW;
  float*       q = out + (size_t)c * HW;
  float s = 0.f, s2 = 0.f;
  for (int i = threadIdx.x; i < HW; i += 256) { float v = p[i]; s += v; s2 += v * v; }
  __shared__ float rs[256], rq[256];
  rs[threadIdx.x] = s; rq[threadIdx.x] = s2;
  __syncthreads();
  for (int o = 128; o > 0; o >>= 1) {
    if (threadIdx.x < o) { rs[threadIdx.x] += rs[threadIdx.x + o]; rq[threadIdx.x] += rq[threadIdx.x + o]; }
    __syncthreads();
  }
  float mean = rs[0] / (float)HW;
  float var  = rq[0] / (float)HW - mean * mean;
  float inv  = rsqrtf(var + 1e-5f);
  for (int i = threadIdx.x; i < HW; i += 256) {
    float v = (p[i] - mean) * inv;
    q[i] = v >= 0.f ? v : 0.2f * v;
  }
}

// ---------------------------------------------------------------------------
// Per-pixel KxK window attention (refine): one wave per pixel, lanes = channels
// ---------------------------------------------------------------------------
template <int KWIN>
__global__ __launch_bounds__(256) void refine_attn(
    const float* __restrict__ q, const float* __restrict__ kk,
    const float* __restrict__ vv, float* __restrict__ out2,
    int C, int H, int W, int pad)
{
  constexpr int T = KWIN * KWIN;
  const int lane = threadIdx.x & 31;
  const int wave = threadIdx.x >> 5;
  const int p = blockIdx.x * 8 + wave;
  const int HW = H * W;
  if (p >= HW) return;
  const int h = p / W, wc = p - h * W;
  const int WP = W + 2 * pad;
  const int HPWP = (H + 2 * pad) * WP;
  const int base = h * WP + wc;
  int off[T];
  #pragma unroll
  for (int t = 0; t < T; ++t) off[t] = (t / KWIN) * WP + (t % KWIN);

  float sim[T];
  #pragma unroll
  for (int t = 0; t < T; ++t) sim[t] = 0.f;
  for (int c = lane; c < C; c += 32) {
    float qv = q[(size_t)c * HW + p];
    const float* kp = kk + (size_t)c * HPWP + base;
    #pragma unroll
    for (int t = 0; t < T; ++t) sim[t] += qv * kp[off[t]];
  }
  #pragma unroll
  for (int t = 0; t < T; ++t) {
    float s = sim[t];
    for (int o = 16; o > 0; o >>= 1) s += __shfl_xor(s, o, 32);
    sim[t] = s;
  }
  float mx = sim[0];
  #pragma unroll
  for (int t = 1; t < T; ++t) mx = fmaxf(mx, sim[t]);
  float den = 0.f;
  #pragma unroll
  for (int t = 0; t < T; ++t) { sim[t] = expf(sim[t] - mx); den += sim[t]; }
  float inv = 1.f / den;
  for (int c = lane; c < C; c += 32) {
    const float* vp = vv + (size_t)c * HPWP + base;
    float s = 0.f;
    #pragma unroll
    for (int t = 0; t < T; ++t) s += sim[t] * vp[off[t]];
    out2[(size_t)c * HW + p] = s * inv;
  }
}

// ---------------------------------------------------------------------------
// Elementwise / resampling helpers
// ---------------------------------------------------------------------------
__global__ __launch_bounds__(256) void prep_input(
    const float* __restrict__ img, const float* __restrict__ mask,
    const float* __restrict__ noise, float* __restrict__ x)
{
  int i = blockIdx.x * 256 + threadIdx.x;
  if (i >= 262144) return;                       // 4*256*256
  x[i] = (i < 196608) ? img[i] + noise[i] * (1.f / 256.f) : mask[i & 65535];
}

__global__ __launch_bounds__(256) void mul_inplace(
    float* __restrict__ a, const float* __restrict__ b, long n)
{
  for (long i = blockIdx.x * 256L + threadIdx.x; i < n; i += (long)gridDim.x * 256L)
    a[i] *= b[i];
}

__global__ __launch_bounds__(256) void combine_k(
    const float* __restrict__ res, const float* __restrict__ o1,
    const float* __restrict__ m, const float* __restrict__ o2,
    float* __restrict__ out, long n)
{
  for (long i = blockIdx.x * 256L + threadIdx.x; i < n; i += (long)gridDim.x * 256L) {
    float mm = m[i];
    out[i] = res[i] + o1[i] * mm + (1.f - mm) * o2[i];
  }
}

__global__ __launch_bounds__(256) void up2x(
    const float* __restrict__ in, float* __restrict__ out, int C, int H, int W)
{
  const int HO = 2 * H, WO = 2 * W;
  const long total = (long)C * HO * WO;
  const float sy = (float)(H - 1) / (float)(HO - 1);
  const float sx = (float)(W - 1) / (float)(WO - 1);
  for (long i = blockIdx.x * 256L + threadIdx.x; i < total; i += (long)gridDim.x * 256L) {
    int wo = (int)(i % WO); long t = i / WO;
    int ho = (int)(t % HO); int c = (int)(t / HO);
    float fy = ho * sy;  int y0 = (int)floorf(fy); int y1 = min(y0 + 1, H - 1); float wy = fy - y0;
    float fx = wo * sx;  int x0 = (int)floorf(fx); int x1 = min(x0 + 1, W - 1); float wx = fx - x0;
    const float* p = in + (size_t)c * H * W;
    float v = (p[y0 * W + x0] * (1.f - wx) + p[y0 * W + x1] * wx) * (1.f - wy)
            + (p[y1 * W + x0] * (1.f - wx) + p[y1 * W + x1] * wx) * wy;
    out[i] = v;
  }
}

__global__ __launch_bounds__(256) void reflect_pad3(
    const float* __restrict__ in, float* __restrict__ out, int C, int H, int W)
{
  const int HP = H + 6, WP = W + 6;
  const long total = (long)C * HP * WP;
  for (long i = blockIdx.x * 256L + threadIdx.x; i < total; i += (long)gridDim.x * 256L) {
    int wo = (int)(i % WP); long t = i / WP;
    int ho = (int)(t % HP); int c = (int)(t / HP);
    int hi = ho - 3, wi = wo - 3;
    hi = hi < 0 ? -hi : (hi >= H ? 2 * H - 2 - hi : hi);
    wi = wi < 0 ? -wi : (wi >= W ? 2 * W - 2 - wi : wi);
    out[i] = in[((size_t)c * H + hi) * W + wi];
  }
}

// ---------------------------------------------------------------------------
// Host orchestration
// ---------------------------------------------------------------------------
extern "C" void kernel_launch(void* const* d_in, const int* in_sizes, int n_in,
                              void* d_out, int out_size, void* d_ws, size_t ws_size,
                              hipStream_t stream)
{
  (void)in_sizes; (void)n_in; (void)out_size;

  const size_t BIG  = 8388608;   // 128*256*256
  const size_t MED  = 4194304;   // 64*256*256
  const size_t PADS = 4326400;   // 64*260*260
  const size_t NEED = 3 * BIG + 5 * MED + 2 * PADS;     // 54,790,144 floats
  const size_t WCAP = 16400000;                         // f16 weight arena (halves)
  if (ws_size < NEED * sizeof(float) + WCAP * sizeof(_Float16)) return;

  float* ws = (float*)d_ws;
  float* X[2] = { ws, ws + BIG };
  float* NORM = ws + 2 * BIG;                 // pre-norm x; reused for out2 & reflect-pad
  float* RES  = NORM + BIG;
  float* Tb   = RES + MED;
  float* OUT1 = Tb + MED;
  float* Mb   = OUT1 + MED;
  float* Qb   = Mb + MED;
  float* Kb   = Qb + MED;
  float* Vb   = Kb + PADS;
  _Float16* wf16base = (_Float16*)(ws + NEED);
  size_t woff = 0;

  const float* img   = (const float*)d_in[0];
  const float* mask  = (const float*)d_in[1];
  const float* noise = (const float*)d_in[2];

  struct BC { int ic, oc, k, s, d, pad, ipad, n0, up; };
  const BC B[14] = {
    {  4,  64, 5, 1, 1, 2, 2, 0, 0 },  // down0
    { 64, 128, 3, 2, 1, 1, 1, 1, 0 },  // down1
    {128, 128, 3, 1, 1, 1, 1, 1, 0 },  // down11
    {128, 256, 3, 2, 1, 1, 1, 1, 0 },  // down2
    {256, 256, 3, 1, 1, 1, 1, 1, 0 },  // middle1
    {256, 256, 3, 1, 2, 2, 1, 1, 0 },  // middle2
    {256, 256, 3, 1, 3, 3, 1, 1, 0 },  // middle3
    {256, 256, 3, 1, 2, 2, 1, 1, 0 },  // middle4
    {256, 256, 3, 1, 3, 3, 1, 1, 0 },  // middle5
    {256, 256, 3, 1, 1, 1, 1, 1, 0 },  // middle6
    {256, 128, 3, 1, 1, 1, 1, 1, 1 },  // up1
    {128, 128, 3, 1, 1, 1, 1, 1, 0 },  // up11
    {128,  64, 3, 1, 1, 1, 1, 1, 1 },  // up2
    { 64,  64, 3, 1, 1, 1, 1, 1, 0 },  // up21
  };

  // repack fp32 weights -> padded f16, then launch templated implicit-GEMM conv
  auto conv = [&](const float* in, int IC, int IH, int IW,
                  const float* wsrc, const float* b, float* out,
                  int OC, int OH, int OW, int Kk, int s, int pad, int dil, int act) {
    int Ktot  = IC * Kk * Kk;
    int KtotP = (Ktot + 63) & ~63;
    int OCP   = (OC + 63) & ~63;
    _Float16* wf = wf16base + woff;
    long totalP = (long)OCP * KtotP;
    woff += (size_t)totalP;
    cvt_w<<<2048, 256, 0, stream>>>(wsrc, wf, OC, Ktot, KtotP, totalP);
    dim3 g((OC + 63) / 64, (OH * OW + 63) / 64);
    switch (Kk) {
      case 1:  conv_wmma_t<1><<<g, 256, 0, stream>>>(in, wf, b, out, IC, IH, IW, OC, OH, OW, KtotP, s, pad, dil, act); break;
      case 3:  conv_wmma_t<3><<<g, 256, 0, stream>>>(in, wf, b, out, IC, IH, IW, OC, OH, OW, KtotP, s, pad, dil, act); break;
      case 5:  conv_wmma_t<5><<<g, 256, 0, stream>>>(in, wf, b, out, IC, IH, IW, OC, OH, OW, KtotP, s, pad, dil, act); break;
      default: conv_wmma_t<7><<<g, 256, 0, stream>>>(in, wf, b, out, IC, IH, IW, OC, OH, OW, KtotP, s, pad, dil, act); break;
    }
  };

  prep_input<<<1024, 256, 0, stream>>>(img, mask, noise, X[0]);

  int cur = 0, H = 256, W = 256;
  int wi = 3;   // inputs flattened in setup_inputs() insertion order

  for (int bi = 0; bi < 14; ++bi) {
    BC c = B[bi];
    const float* w1 = (const float*)d_in[wi++]; const float* b1 = (const float*)d_in[wi++];
    const float* w2 = (const float*)d_in[wi++]; const float* b2 = (const float*)d_in[wi++];
    const float* wm = (const float*)d_in[wi++]; const float* bm = (const float*)d_in[wi++];
    const float* wq = (const float*)d_in[wi++]; const float* bq = (const float*)d_in[wi++];
    const float* wk = (const float*)d_in[wi++]; const float* bk = (const float*)d_in[wi++];
    const float* wv = (const float*)d_in[wi++]; const float* bv = (const float*)d_in[wi++];
    const float* wp = nullptr; const float* bp = nullptr;
    bool proj = (c.ic != c.oc) || (c.s != 1);
    if (proj) { wp = (const float*)d_in[wi++]; bp = (const float*)d_in[wi++]; }

    if (c.up) {
      up2x<<<4096, 256, 0, stream>>>(X[cur], X[1 - cur], c.ic, H, W);
      cur ^= 1; H *= 2; W *= 2;
    }
    const int OH = (H + 2 * c.pad - c.d * (c.k - 1) - 1) / c.s + 1;
    const int OW = (W + 2 * c.pad - c.d * (c.k - 1) - 1) / c.s + 1;
    const float* xin = X[cur];

    const float* resPtr;
    if (proj) { conv(xin, c.ic, H, W, wp, bp, RES, c.oc, OH, OW, 1, c.s, 0, 1, 0); resPtr = RES; }
    else      { resPtr = xin; }

    const float* t_in = xin;
    if (c.n0) { inorm_lrelu<<<c.ic, 256, 0, stream>>>(xin, NORM, H * W); t_in = NORM; }

    conv(t_in, c.ic, H, W, w1, b1, Tb, c.oc, OH, OW, c.k, c.s, c.pad, c.d, 0);
    inorm_lrelu<<<c.oc, 256, 0, stream>>>(Tb, Tb, OH * OW);                 // in-place

    conv(Tb, c.oc, OH, OW, w2, b2, OUT1, c.oc, OH, OW, c.k, 1, c.ipad, 1, 0);
    conv(Tb, c.oc, OH, OW, wm, bm, Mb,   c.oc, OH, OW, c.k, 1, c.ipad, 1, 1); // +sigmoid

    long n_t = (long)c.oc * OH * OW;
    mul_inplace<<<4096, 256, 0, stream>>>(Tb, Mb, n_t);                     // Tb = x*m

    conv(Tb, c.oc, OH, OW, wq, bq, Qb, c.oc, OH, OW, 1, 1, 0, 1, 0);
    const int HP = OH + 2 * c.ipad, WPd = OW + 2 * c.ipad;                  // padded k/v maps
    conv(Tb, c.oc, OH, OW, wk, bk, Kb, c.oc, HP, WPd, 1, 1, c.ipad, 1, 0);
    conv(Tb, c.oc, OH, OW, wv, bv, Vb, c.oc, HP, WPd, 1, 1, c.ipad, 1, 0);

    float* O2 = NORM;                                                       // NORM dead here
    dim3 ag((OH * OW + 7) / 8);
    if (c.k == 3) refine_attn<3><<<ag, 256, 0, stream>>>(Qb, Kb, Vb, O2, c.oc, OH, OW, c.ipad);
    else          refine_attn<5><<<ag, 256, 0, stream>>>(Qb, Kb, Vb, O2, c.oc, OH, OW, c.ipad);

    combine_k<<<4096, 256, 0, stream>>>(resPtr, OUT1, Mb, O2, X[1 - cur], n_t);
    cur ^= 1; H = OH; W = OW;
  }

  // Final: reflect pad 3 -> 7x7 conv (64->3) -> tanh, straight into d_out
  reflect_pad3<<<4096, 256, 0, stream>>>(X[cur], NORM, 64, H, W);
  const float* ow = (const float*)d_in[wi];
  const float* ob = (const float*)d_in[wi + 1];
  conv(NORM, 64, H + 6, W + 6, ow, ob, (float*)d_out, 3, H, W, 7, 1, 0, 1, 2);
}
